// RNNgainBeta_2095944040496
// MI455X (gfx1250) — compile-verified, matrix operand order checked by reading, not run on hardware
//
#include <hip/hip_runtime.h>
#include <hip/hip_bf16.h>

typedef __attribute__((ext_vector_type(16))) __bf16 v16bf;
typedef __attribute__((ext_vector_type(8)))  __bf16 v8bf;
typedef __attribute__((ext_vector_type(8)))  float  v8f;

#define Hdim 1024
#define Bdim 64
#define Tdim 500
#define Idim 3
#define Odim 2
#define NBLOCKS 32
#define WPB 8          // waves per block (wave32)
#define THRES 30.0f
#define NOISE_STD 0.05f
#define ALPHA 0.2f

// ---------------- prep: W2[j][k] = relu(wrec*refEI)*refEI * mwrec * relu(g[k])  (bf16)
__global__ void prep_w2(const float* __restrict__ wrec, const float* __restrict__ refEI,
                        const float* __restrict__ mwrec, const float* __restrict__ g,
                        __bf16* __restrict__ W2) {
  int j = blockIdx.x;
  for (int k = threadIdx.x; k < Hdim; k += blockDim.x) {
    float s  = refEI[j * Hdim + k];
    float w  = wrec[j * Hdim + k];
    float e  = fmaxf(w * s, 0.0f) * s * mwrec[j * Hdim + k];
    float gk = fmaxf(g[k], 0.0f);
    W2[j * Hdim + k] = (__bf16)(e * gk);
  }
}

// ---------------- prep: h=h0 broadcast, r0 = beta-softplus(h0+b) (mask on beta*r_pre),
//                  wout2 = relu(g)*wout, zero barrier counters
__global__ void prep_state(const float* __restrict__ h0, const float* __restrict__ bvec,
                           const float* __restrict__ g, const float* __restrict__ wout,
                           const float* __restrict__ beta_p,
                           float* __restrict__ hbuf, __bf16* __restrict__ rg0,
                           float* __restrict__ wout2, int* __restrict__ cnt) {
  int idx = blockIdx.x * blockDim.x + threadIdx.x;
  float beta = beta_p[0];
  if (idx < Bdim * Hdim) {
    int hh = idx & (Hdim - 1);
    float h0v = h0[hh];
    hbuf[idx] = h0v;
    float rp = h0v + bvec[hh];
    float r = (beta * rp < THRES) ? (__logf(1.0f + __expf(beta * rp)) / beta) : rp;
    rg0[idx] = (__bf16)r;
  }
  if (idx < Hdim * Odim) {
    int hh = idx >> 1, o = idx & 1;
    wout2[idx] = fmaxf(g[hh], 0.0f) * wout[hh * Odim + o];
  }
  if (idx < 512) cnt[idx] = 0;
}

__global__ void zero_out(float* __restrict__ out, int n) {
  int idx = blockIdx.x * blockDim.x + threadIdx.x;
  if (idx < n) out[idx] = 0.0f;
}

// ---------------- persistent RNN: one 16x16 output tile per wave, WMMA bf16 GEMM,
//                  grid barrier per step, double-buffered r
__global__ __launch_bounds__(256) void rnn_persistent(
    const float* __restrict__ input, const float* __restrict__ wi,
    const float* __restrict__ bvec, const float* __restrict__ beta_p,
    const float* __restrict__ noise, const __bf16* __restrict__ W2,
    __bf16* __restrict__ rbuf0, __bf16* __restrict__ rbuf1,
    float* __restrict__ hbuf, const float* __restrict__ wout2,
    float* __restrict__ out, int* __restrict__ cnt) {
  const int tid  = threadIdx.x;
  const int wave = tid >> 5;
  const int lane = tid & 31;
  const int gw   = blockIdx.x * WPB + wave;   // 0..255
  const int mt   = gw & 3;                    // batch tile  (4)
  const int nt   = gw >> 2;                   // hidden tile (64)
  const int lm   = lane & 15;
  const int hf   = lane >> 4;
  const float beta = beta_p[0];

  const int n = nt * 16 + lm;                 // hidden index for C-layout columns
  const float w0 = wout2[n * Odim + 0];
  const float w1 = wout2[n * Odim + 1];
  const float bn = bvec[n];
  const float wi0 = wi[0 * Hdim + n];
  const float wi1 = wi[1 * Hdim + n];
  const float wi2 = wi[2 * Hdim + n];

  // t=0 readout from r0
  {
#pragma unroll
    for (int v = 0; v < 8; ++v) {
      int m = mt * 16 + v + 8 * hf;
      float r = (float)rbuf0[m * Hdim + n];
      float p0 = r * w0, p1 = r * w1;
      for (int d = 1; d < 16; d <<= 1) { p0 += __shfl_xor(p0, d, 32); p1 += __shfl_xor(p1, d, 32); }
      if (lm == 0) {
        atomicAdd(&out[(m * Tdim + 0) * Odim + 0], p0);
        atomicAdd(&out[(m * Tdim + 0) * Odim + 1], p1);
      }
    }
  }

  const __bf16* rprev = rbuf0;
  __bf16* rnext = rbuf1;

  for (int s = 1; s < Tdim; ++s) {
    // --- tile GEMM: acc = r_prev[mt rows] @ W2^T[nt cols], K = 1024
    v8f acc = {};
    const __bf16* arow = rprev + (mt * 16 + lm) * Hdim + 8 * hf;
    const __bf16* brow = W2    + (nt * 16 + lm) * Hdim + 8 * hf;
#pragma unroll 4
    for (int k0 = 0; k0 < Hdim; k0 += 32) {
      v8bf a0 = *(const v8bf*)(arow + k0);
      v8bf a1 = *(const v8bf*)(arow + k0 + 16);
      v8bf b0 = *(const v8bf*)(brow + k0);
      v8bf b1 = *(const v8bf*)(brow + k0 + 16);
      v16bf A, Bm;
#pragma unroll
      for (int i = 0; i < 8; ++i) { A[i] = a0[i]; A[i + 8] = a1[i]; Bm[i] = b0[i]; Bm[i + 8] = b1[i]; }
      acc = __builtin_amdgcn_wmma_f32_16x16x32_bf16(false, A, false, Bm, (short)0, acc,
                                                    false, false);
    }

    // --- elementwise state update + activation + readout partials
#pragma unroll
    for (int v = 0; v < 8; ++v) {
      int m = mt * 16 + v + 8 * hf;
      const float* xp = input + (m * Tdim + (s - 1)) * Idim;
      float xw = xp[0] * wi0 + xp[1] * wi1 + xp[2] * wi2;
      float nz = noise[(size_t)(m * Tdim + (s - 1)) * Hdim + n];
      float ho = hbuf[m * Hdim + n];
      float hn = ho + NOISE_STD * nz + ALPHA * (-ho + acc[v] + xw);
      hbuf[m * Hdim + n] = hn;
      float rp = hn + bn;
      float r = (rp < THRES) ? (__logf(1.0f + __expf(beta * rp)) / beta) : rp;
      rnext[m * Hdim + n] = (__bf16)r;
      float p0 = r * w0, p1 = r * w1;
      for (int d = 1; d < 16; d <<= 1) { p0 += __shfl_xor(p0, d, 32); p1 += __shfl_xor(p1, d, 32); }
      if (lm == 0) {
        atomicAdd(&out[(m * Tdim + s) * Odim + 0], p0);
        atomicAdd(&out[(m * Tdim + s) * Odim + 1], p1);
      }
    }

    // --- grid barrier (release writes of rnext, then all blocks arrive)
    __syncthreads();
    if (tid == 0) {
      __threadfence();
      atomicAdd(&cnt[s], 1);
      while (__hip_atomic_load(&cnt[s], __ATOMIC_ACQUIRE, __HIP_MEMORY_SCOPE_AGENT) < NBLOCKS) {}
    }
    __syncthreads();

    const __bf16* t = rprev; rprev = rnext; rnext = (__bf16*)t;
  }
}

extern "C" void kernel_launch(void* const* d_in, const int* in_sizes, int n_in,
                              void* d_out, int out_size, void* d_ws, size_t ws_size,
                              hipStream_t stream) {
  const float* input = (const float*)d_in[0];
  const float* wi    = (const float*)d_in[1];
  const float* wrec  = (const float*)d_in[2];
  const float* wout  = (const float*)d_in[3];
  const float* bvec  = (const float*)d_in[4];
  const float* gv    = (const float*)d_in[5];
  const float* mwrec = (const float*)d_in[6];
  const float* refEI = (const float*)d_in[7];
  const float* h0    = (const float*)d_in[8];
  const float* beta  = (const float*)d_in[9];
  const float* noise = (const float*)d_in[10];
  float* out = (float*)d_out;

  char* ws = (char*)d_ws;
  __bf16* W2    = (__bf16*)(ws);                        // 2 MB
  __bf16* r0    = (__bf16*)(ws + 2097152);              // 128 KB
  __bf16* r1    = (__bf16*)(ws + 2097152 + 131072);     // 128 KB
  float*  hbuf  = (float*)(ws + 2359296);               // 256 KB
  float*  wout2 = (float*)(ws + 2621440);               // 8 KB
  int*    cnt   = (int*)(ws + 2629632);                 // 2 KB

  prep_w2<<<Hdim, 256, 0, stream>>>(wrec, refEI, mwrec, gv, W2);
  prep_state<<<(Bdim * Hdim + 255) / 256, 256, 0, stream>>>(h0, bvec, gv, wout, beta,
                                                            hbuf, r0, wout2, cnt);
  zero_out<<<(Bdim * Tdim * Odim + 255) / 256, 256, 0, stream>>>(out, Bdim * Tdim * Odim);
  rnn_persistent<<<NBLOCKS, NBLOCKS * WPB, 0, stream>>>(input, wi, bvec, beta, noise, W2,
                                                        r0, r1, hbuf, wout2, out, cnt);
}
